// TransformerAttention_43224550867315
// MI455X (gfx1250) — matrix operand order repaired and checked by reading, not compiled, not run on hardware
//
#include <hip/hip_runtime.h>
#include <hip/hip_bf16.h>

typedef __attribute__((ext_vector_type(16))) __bf16 v16bf;
typedef __attribute__((ext_vector_type(8)))  float  v8f;
typedef __attribute__((ext_vector_type(4)))  unsigned int u32x4;
typedef __attribute__((ext_vector_type(4)))  int  i32x4;
typedef __attribute__((ext_vector_type(8)))  int  i32x8;

#define NTOK 8192
#define HDIM 2048
#define TSEQ 2048
#define NH   16
#define HD   128

union FragU { uint4 u[2]; v16bf v; };

__device__ __forceinline__ v16bf ld_frag2(const __bf16* p0, const __bf16* p1) {
    FragU f;
    f.u[0] = *(const uint4*)p0;
    f.u[1] = *(const uint4*)p1;
    return f.v;
}

__device__ __forceinline__ v8f wmma_bf16(v16bf a, v16bf b, v8f c) {
    return __builtin_amdgcn_wmma_f32_16x16x32_bf16(false, a, false, b, (short)0, c, false, false);
}

// ---------------------------------------------------------------------------
// Kernel 1: QKV projection GEMM.  out = x @ W^T, epilogue -> [B,nH,T,hd] bf16
// grid: (HDIM/64, NTOK/128, 3)  block: 256 (8 waves).  K-chunk = 64.
// ---------------------------------------------------------------------------
__global__ __launch_bounds__(256) void qkv_gemm_kernel(
    const float* __restrict__ X,
    const float* __restrict__ Wq, const float* __restrict__ Wk, const float* __restrict__ Wv,
    __bf16* __restrict__ Qg, __bf16* __restrict__ Kg, __bf16* __restrict__ Vg)
{
    constexpr int LDT = 72;                       // padded LDS stride (halves), 144B = 9*16B
    __shared__ __bf16 Xs[128 * LDT];
    __shared__ __bf16 Ws[64 * LDT];

    const int z = blockIdx.z;
    const float* W   = (z == 0) ? Wq : (z == 1) ? Wk : Wv;
    __bf16*      Out = (z == 0) ? Qg : (z == 1) ? Kg : Vg;

    const int tid  = threadIdx.x;
    const int lane = tid & 31, w = tid >> 5;
    const int ln   = lane & 15, sel = lane >> 4;
    const int m0 = blockIdx.y * 128;
    const int n0 = blockIdx.x * 64;
    const int wm = (w >> 1) * 32, wn = (w & 1) * 32;

    v8f acc[4] = {};

    for (int kc = 0; kc < HDIM; kc += 64) {
        __syncthreads();
        // stage X tile (128x64 f32 -> bf16)
        #pragma unroll
        for (int i = 0; i < 8; i++) {
            int idx = tid + i * 256, r = idx >> 4, c = idx & 15;
            float4 f = *(const float4*)(X + (size_t)(m0 + r) * HDIM + kc + c * 4);
            __bf16* d = Xs + r * LDT + c * 4;
            d[0] = (__bf16)f.x; d[1] = (__bf16)f.y; d[2] = (__bf16)f.z; d[3] = (__bf16)f.w;
        }
        // stage W tile (64x64 f32 -> bf16)
        #pragma unroll
        for (int i = 0; i < 4; i++) {
            int idx = tid + i * 256, r = idx >> 4, c = idx & 15;
            float4 f = *(const float4*)(W + (size_t)(n0 + r) * HDIM + kc + c * 4);
            __bf16* d = Ws + r * LDT + c * 4;
            d[0] = (__bf16)f.x; d[1] = (__bf16)f.y; d[2] = (__bf16)f.z; d[3] = (__bf16)f.w;
        }
        __syncthreads();

        #pragma unroll
        for (int kk = 0; kk < 64; kk += 32) {
            // A fragments (16x32): lane row = m, halves K 0-7/16-23 (lo) or 8-15/24-31 (hi)
            const __bf16* pa0 = Xs + (wm + ln) * LDT + kk + sel * 8;
            const __bf16* pa1 = Xs + (wm + 16 + ln) * LDT + kk + sel * 8;
            v16bf a0 = ld_frag2(pa0, pa0 + 16);
            v16bf a1 = ld_frag2(pa1, pa1 + 16);
            // B fragments (32x16): lane col = n, halves K 0-7/16-23 (lo) or 8-15/24-31 (hi)
            const __bf16* pb0 = Ws + (wn + ln) * LDT + kk + sel * 8;
            const __bf16* pb1 = Ws + (wn + 16 + ln) * LDT + kk + sel * 8;
            v16bf b0 = ld_frag2(pb0, pb0 + 16);
            v16bf b1 = ld_frag2(pb1, pb1 + 16);

            acc[0] = wmma_bf16(a0, b0, acc[0]);
            acc[1] = wmma_bf16(a0, b1, acc[1]);
            acc[2] = wmma_bf16(a1, b0, acc[2]);
            acc[3] = wmma_bf16(a1, b1, acc[3]);
        }
    }

    // epilogue: C layout lane(l): M = r + 8*(l>=16), N = l&15 ; write head-split bf16
    #pragma unroll
    for (int t2 = 0; t2 < 4; t2++) {
        int tm = t2 >> 1, tn = t2 & 1;
        #pragma unroll
        for (int j = 0; j < 8; j++) {
            int tok = m0 + wm + tm * 16 + j + sel * 8;
            int h   = n0 + wn + tn * 16 + ln;
            int b = tok >> 11, t = tok & 2047;
            int head = h >> 7, d = h & 127;
            Out[((size_t)((b * NH + head) * TSEQ + t)) * HD + d] = (__bf16)acc[t2][j];
        }
    }
}

// ---------------------------------------------------------------------------
// Kernel 2: RoPE in-place on Q and K.  grid: (BHT*64/256, 1, 2)
// ---------------------------------------------------------------------------
__global__ __launch_bounds__(256) void rope_kernel(__bf16* __restrict__ Qg, __bf16* __restrict__ Kg)
{
    __bf16* P = blockIdx.z ? Kg : Qg;
    int idx = blockIdx.x * 256 + threadIdx.x;     // [0, 4*16*2048*64)
    int d   = idx & 63;
    int bht = idx >> 6;
    int t   = bht & (TSEQ - 1);
    size_t base = (size_t)bht * HD;

    float ang = (float)t * __powf(10000.0f, -(float)d * (1.0f / 64.0f));
    float s, c;
    __sincosf(ang, &s, &c);
    float q0 = (float)P[base + d];
    float q1 = (float)P[base + d + 64];
    P[base + d]      = (__bf16)(q0 * c - q1 * s);
    P[base + d + 64] = (__bf16)(q1 * c + q0 * s);
}

// ---------------------------------------------------------------------------
// Kernel 3: causal flash attention.  grid: (T/128, nH, B)  block: 256
// K tile staged by the Tensor Data Mover (TDM) when available; V tile
// transposed by VALU into LDS; per-wave online softmax; WMMA everywhere.
// ---------------------------------------------------------------------------
__global__ __launch_bounds__(256) void attn_kernel(
    const __bf16* __restrict__ Qg, const __bf16* __restrict__ Kg,
    const __bf16* __restrict__ Vg, __bf16* __restrict__ AO)
{
    constexpr int LDK = 136;   // 64x128 K tile, padded rows: 256B data + 16B pad
    constexpr int LDV = 72;    // 128x64 V^T tile, padded (144B rows)
    constexpr int LDP = 72;    // per-wave 16x64 P slab
    __shared__ __bf16 Ks[64 * LDK];
    __shared__ __bf16 Vts[128 * LDV];
    __shared__ __bf16 Pws[8 * 16 * LDP];

    const int tid = threadIdx.x;
    const int lane = tid & 31, w = tid >> 5;
    const int ln = lane & 15, sel = lane >> 4;
    const int qb = blockIdx.x, h = blockIdx.y, b = blockIdx.z;

    const size_t headoff = (size_t)(b * NH + h) * TSEQ * HD;
    const __bf16* Qh = Qg + headoff;
    const __bf16* Kh = Kg + headoff;
    const __bf16* Vh = Vg + headoff;

    // preload Q fragments for this wave's 16 rows, all 4 K-chunks (A layout)
    const int qrow = qb * 128 + w * 16 + ln;
    v16bf qf[4];
    #pragma unroll
    for (int kc = 0; kc < 4; kc++) {
        const __bf16* p = Qh + (size_t)qrow * HD + kc * 32 + sel * 8;
        qf[kc] = ld_frag2(p, p + 16);
    }

    v8f oacc[8] = {};
    float rmax[8], rsum[8];
    #pragma unroll
    for (int j = 0; j < 8; j++) { rmax[j] = -1e30f; rsum[j] = 0.0f; }

    const float scale = 0.08838834764831845f;     // 1/sqrt(128)
    const int ktend = qb * 2 + 1;

    for (int kt = 0; kt <= ktend; kt++) {
        __syncthreads();

#if __has_builtin(__builtin_amdgcn_tensor_load_to_lds)
        // ---- K tile via Tensor Data Mover: 64 rows x 256B, +16B LDS pad/row ----
        if (w == 0) {
            unsigned ldsoff = (unsigned)(size_t)(void*)Ks;
            unsigned long long ga =
                (unsigned long long)(size_t)(Kh + (size_t)(kt * 64) * HD);
            u32x4 g0;
            g0.x = 1u;                                        // count=1, user mode
            g0.y = ldsoff;                                    // lds_addr
            g0.z = (unsigned)(ga & 0xffffffffu);              // global_addr[31:0]
            g0.w = (unsigned)((ga >> 32) & 0x1ffffffu) | (2u << 30); // addr[56:32], type=2
            i32x8 g1;
            g1[0] = (1 << 16) | (1 << 20) | (5 << 22) | (3 << 25);
            //      data_size=2B | pad_enable | pad_interval=64dw | pad_amount=4dw
            g1[1] = (HD << 16);          // tensor_dim0[15:0]=128 (atomic_addr=0)
            g1[2] = (64 << 16);          // tensor_dim0 hi=0, tensor_dim1[15:0]=64
            g1[3] = (HD << 16);          // tensor_dim1 hi=0, tile_dim0=128
            g1[4] = 64;                  // tile_dim1=64, tile_dim2=0
            g1[5] = HD;                  // tensor_dim0_stride low = 128 elements
            g1[6] = 0;
            g1[7] = 0;
            i32x4 z4 = {};
#if defined(__clang_major__) && __clang_major__ >= 23
            i32x8 z8 = {};
            __builtin_amdgcn_tensor_load_to_lds(g0, g1, z4, z4, z8, 0);
#else
            __builtin_amdgcn_tensor_load_to_lds(g0, g1, z4, z4, 0);
#endif
        }
#else
        // fallback: cooperative vectorized K staging
        #pragma unroll
        for (int i = 0; i < 4; i++) {
            int idx = tid + i * 256, r = idx >> 4, c = idx & 15;
            uint4 u = *(const uint4*)(Kh + (size_t)(kt * 64 + r) * HD + c * 8);
            *(uint4*)(Ks + r * LDK + c * 8) = u;
        }
#endif
        // stage V tile transposed: Vts[d][key]   (overlaps with TDM transfer)
        #pragma unroll
        for (int i = 0; i < 4; i++) {
            int idx = tid + i * 256, r = idx >> 4, c = idx & 15;
            union { uint4 u; __bf16 hh[8]; } uu;
            uu.u = *(const uint4*)(Vh + (size_t)(kt * 64 + r) * HD + c * 8);
            #pragma unroll
            for (int e = 0; e < 8; e++) Vts[(c * 8 + e) * LDV + r] = uu.hh[e];
        }
        if (kt < ktend) {   // prefetch next V tile into cache
            __builtin_prefetch(Vh + (size_t)((kt + 1) * 64) * HD + tid * 32, 0, 0);
        }
#if __has_builtin(__builtin_amdgcn_tensor_load_to_lds)
        if (w == 0) __builtin_amdgcn_s_wait_tensorcnt(0);
#endif
        __syncthreads();

        // S = Q * K^T : 4 n-tiles x 4 k-chunks of WMMA
        v8f sa[4] = {};
        #pragma unroll
        for (int kc = 0; kc < 4; kc++) {
            #pragma unroll
            for (int nt = 0; nt < 4; nt++) {
                const __bf16* p = Ks + (nt * 16 + ln) * LDK + kc * 32 + sel * 8;
                v16bf bf = ld_frag2(p, p + 16);
                sa[nt] = wmma_bf16(qf[kc], bf, sa[nt]);
            }
        }

        // causal mask + online softmax (row stats per lane across the 16-lane half)
        const int qg0 = qb * 128 + w * 16 + sel * 8;
        const int kb0 = kt * 64 + ln;
        #pragma unroll
        for (int j = 0; j < 8; j++) {
            int q = qg0 + j;
            float tmax = -1e30f;
            #pragma unroll
            for (int nt = 0; nt < 4; nt++) {
                float v = sa[nt][j] * scale;
                if (kb0 + nt * 16 > q) v = -1e30f;
                sa[nt][j] = v;
                tmax = fmaxf(tmax, v);
            }
            #pragma unroll
            for (int off = 1; off < 16; off <<= 1) tmax = fmaxf(tmax, __shfl_xor(tmax, off));
            float mnew = fmaxf(rmax[j], tmax);
            float corr = __expf(rmax[j] - mnew);
            rmax[j] = mnew;
            float ls = 0.0f;
            #pragma unroll
            for (int nt = 0; nt < 4; nt++) {
                float p = __expf(sa[nt][j] - mnew);
                sa[nt][j] = p;
                ls += p;
            }
            #pragma unroll
            for (int off = 1; off < 16; off <<= 1) ls += __shfl_xor(ls, off);
            rsum[j] = rsum[j] * corr + ls;
            #pragma unroll
            for (int dt = 0; dt < 8; dt++) oacc[dt][j] *= corr;
        }

        // C-layout -> A-layout via per-wave LDS slab
        __bf16* Pw = Pws + w * 16 * LDP;
        #pragma unroll
        for (int j = 0; j < 8; j++) {
            int m = j + sel * 8;
            #pragma unroll
            for (int nt = 0; nt < 4; nt++)
                Pw[m * LDP + nt * 16 + ln] = (__bf16)sa[nt][j];
        }
        asm volatile("s_wait_dscnt 0x0" ::: "memory");  // per-wave DS in order; fence compiler

        // O += P * V
        #pragma unroll
        for (int kc = 0; kc < 2; kc++) {
            const __bf16* pa = Pw + ln * LDP + kc * 32 + sel * 8;
            v16bf af = ld_frag2(pa, pa + 16);
            #pragma unroll
            for (int dt = 0; dt < 8; dt++) {
                const __bf16* pb = Vts + (dt * 16 + ln) * LDV + kc * 32 + sel * 8;
                v16bf bf = ld_frag2(pb, pb + 16);
                oacc[dt] = wmma_bf16(af, bf, oacc[dt]);
            }
        }
    }

    // normalize and write merged-head layout [B, T, H] bf16
    #pragma unroll
    for (int dt = 0; dt < 8; dt++) {
        #pragma unroll
        for (int j = 0; j < 8; j++) {
            int q = qb * 128 + w * 16 + j + sel * 8;
            int d = dt * 16 + ln;
            AO[((size_t)(b * TSEQ + q)) * HDIM + h * HD + d] = (__bf16)(oacc[dt][j] / rsum[j]);
        }
    }
}

// ---------------------------------------------------------------------------
// Kernel 4: output projection.  out = AO @ Wo^T, f32 result.  K-chunk = 64.
// grid: (HDIM/64, NTOK/128)  block: 256
// ---------------------------------------------------------------------------
__global__ __launch_bounds__(256) void out_gemm_kernel(
    const __bf16* __restrict__ AO, const float* __restrict__ Wo, float* __restrict__ Out)
{
    constexpr int LDT = 72;
    __shared__ __bf16 Xs[128 * LDT];
    __shared__ __bf16 Ws[64 * LDT];

    const int tid = threadIdx.x;
    const int lane = tid & 31, w = tid >> 5;
    const int ln = lane & 15, sel = lane >> 4;
    const int m0 = blockIdx.y * 128;
    const int n0 = blockIdx.x * 64;
    const int wm = (w >> 1) * 32, wn = (w & 1) * 32;

    v8f acc[4] = {};

    for (int kc = 0; kc < HDIM; kc += 64) {
        __syncthreads();
        // stage AO tile (already bf16, straight b128 copies): 128x64 halves
        #pragma unroll
        for (int i = 0; i < 4; i++) {
            int idx = tid + i * 256, r = idx >> 3, c = idx & 7;
            uint4 u = *(const uint4*)(AO + (size_t)(m0 + r) * HDIM + kc + c * 8);
            *(uint4*)(Xs + r * LDT + c * 8) = u;
        }
        // stage Wo tile (64x64 f32 -> bf16)
        #pragma unroll
        for (int i = 0; i < 4; i++) {
            int idx = tid + i * 256, r = idx >> 4, c = idx & 15;
            float4 f = *(const float4*)(Wo + (size_t)(n0 + r) * HDIM + kc + c * 4);
            __bf16* d = Ws + r * LDT + c * 4;
            d[0] = (__bf16)f.x; d[1] = (__bf16)f.y; d[2] = (__bf16)f.z; d[3] = (__bf16)f.w;
        }
        __syncthreads();

        #pragma unroll
        for (int kk = 0; kk < 64; kk += 32) {
            const __bf16* pa0 = Xs + (wm + ln) * LDT + kk + sel * 8;
            const __bf16* pa1 = Xs + (wm + 16 + ln) * LDT + kk + sel * 8;
            v16bf a0 = ld_frag2(pa0, pa0 + 16);
            v16bf a1 = ld_frag2(pa1, pa1 + 16);
            const __bf16* pb0 = Ws + (wn + ln) * LDT + kk + sel * 8;
            const __bf16* pb1 = Ws + (wn + 16 + ln) * LDT + kk + sel * 8;
            v16bf b0 = ld_frag2(pb0, pb0 + 16);
            v16bf b1 = ld_frag2(pb1, pb1 + 16);

            acc[0] = wmma_bf16(a0, b0, acc[0]);
            acc[1] = wmma_bf16(a0, b1, acc[1]);
            acc[2] = wmma_bf16(a1, b0, acc[2]);
            acc[3] = wmma_bf16(a1, b1, acc[3]);
        }
    }

    #pragma unroll
    for (int t2 = 0; t2 < 4; t2++) {
        int tm = t2 >> 1, tn = t2 & 1;
        #pragma unroll
        for (int j = 0; j < 8; j++) {
            int tok = m0 + wm + tm * 16 + j + sel * 8;
            int hcol = n0 + wn + tn * 16 + ln;
            Out[(size_t)tok * HDIM + hcol] = acc[t2][j];
        }
    }
}

// ---------------------------------------------------------------------------
extern "C" void kernel_launch(void* const* d_in, const int* in_sizes, int n_in,
                              void* d_out, int out_size, void* d_ws, size_t ws_size,
                              hipStream_t stream) {
    (void)in_sizes; (void)n_in; (void)out_size; (void)ws_size;
    const float* x  = (const float*)d_in[0];
    const float* Wq = (const float*)d_in[1];
    const float* Wk = (const float*)d_in[2];
    const float* Wv = (const float*)d_in[3];
    const float* Wo = (const float*)d_in[4];

    const size_t PER = (size_t)4 * NH * TSEQ * HD;   // 16,777,216 elements
    __bf16* Qg = (__bf16*)d_ws;
    __bf16* Kg = Qg + PER;
    __bf16* Vg = Kg + PER;
    __bf16* AO = Vg + PER;

    qkv_gemm_kernel<<<dim3(HDIM / 64, NTOK / 128, 3), 256, 0, stream>>>(x, Wq, Wk, Wv, Qg, Kg, Vg);
    rope_kernel<<<dim3((4 * NH * TSEQ * 64) / 256, 1, 2), 256, 0, stream>>>(Qg, Kg);
    attn_kernel<<<dim3(TSEQ / 128, NH, 4), 256, 0, stream>>>(Qg, Kg, Vg, AO);
    out_gemm_kernel<<<dim3(HDIM / 64, NTOK / 128, 1), 256, 0, stream>>>(AO, Wo, (float*)d_out);
}
